// DynamicsEnsemble_80676665688822
// MI455X (gfx1250) — compile-verified, matrix operand order checked by reading, not run on hardware
//
#include <hip/hip_runtime.h>
#include <hip/hip_bf16.h>

#if defined(__has_builtin)
#if __has_builtin(__builtin_amdgcn_sched_barrier)
#define SCHED_FENCE() __builtin_amdgcn_sched_barrier(0)
#endif
#endif
#ifndef SCHED_FENCE
#define SCHED_FENCE()
#endif

typedef _Float16 v16h __attribute__((ext_vector_type(16)));
typedef _Float16 v8h  __attribute__((ext_vector_type(8)));
typedef _Float16 v4h  __attribute__((ext_vector_type(4)));
typedef float    v8f  __attribute__((ext_vector_type(8)));
typedef float    v4f  __attribute__((ext_vector_type(4)));

namespace {
constexpr int kE = 8;            // ensemble size
constexpr int kS = 64;           // state dim
constexpr int kA = 16;           // action dim
constexpr int kH = 256;          // hidden dim
constexpr int kG = kS * kA;      // 1024 (g head width)
constexpr int BT = 32;           // batch rows per workgroup (2 WMMA M-tiles)
constexpr int WAVES = 16;
constexpr int THREADS = WAVES * 32;

// Packed-weight fragment bookkeeping. One fragment = 32x16 f16 B-operand tile
// in exact WMMA lane layout: 32 lanes x 16 halves, lane-major (32B per lane).
constexpr size_t FRAG_H = 512;   // halves per fragment (1 KB)
constexpr size_t F1 = (size_t)kE * (kH / 16) * (kS / 32);   //  256 frags
constexpr size_t F2 = (size_t)kE * (kH / 16) * (kH / 32);   // 1024
constexpr size_t Ff = (size_t)kE * (kS / 16) * (kH / 32);   //  256
constexpr size_t Fg = (size_t)kE * (kG / 16) * (kH / 32);   // 4096
constexpr size_t OFF_W1 = 0;
constexpr size_t OFF_W2 = OFF_W1 + F1 * FRAG_H;
constexpr size_t OFF_Wf = OFF_W2 + F2 * FRAG_H;
constexpr size_t OFF_Wg = OFF_Wf + Ff * FRAG_H;
constexpr size_t TOT_H  = OFF_Wg + Fg * FRAG_H;             // total halves
constexpr size_t NEED_WS_BYTES = TOT_H * sizeof(_Float16);  // ~5.5 MB
}

// D = A(16x32 f16) x B(32x16 f16) + C(16x16 f32)
__device__ __forceinline__ v8f wmma_f16(v16h a, v16h b, v8f c) {
    return __builtin_amdgcn_wmma_f32_16x16x32_f16(false, a, false, b, (short)0, c,
                                                  false, false);
}

// A fragment (16x32 f16) from a row-major f16 LDS tile with leading dim ldk.
// CDNA5 16-bit A layout: lanes 0-15 -> row=lane, K 0-7 / 16-23;
//                        lanes 16-31 -> row=lane-16, K 8-15 / 24-31.
__device__ __forceinline__ v16h load_a_lds(const _Float16* base, int ldk, int row0,
                                           int k0, int lane) {
    const int half = lane >> 4;
    const int r = lane & 15;
    const _Float16* p = base + (size_t)(row0 + r) * ldk + k0 + (half ? 8 : 0);
    v8h lo = *(const v8h*)(p);
    v8h hi = *(const v8h*)(p + 16);
    v16h a;
#pragma unroll
    for (int i = 0; i < 8; ++i) { a[i] = lo[i]; a[8 + i] = hi[i]; }
    return a;
}

// B fragment from pre-packed f16 fragments: lane's 16 halves are contiguous 32B,
// whole wave reads one contiguous 1 KB block -> two coalesced b128 loads/lane.
__device__ __forceinline__ v16h load_b_packed(const _Float16* __restrict__ wp,
                                              size_t fragIdx, int lane) {
    const _Float16* p = wp + fragIdx * FRAG_H + (size_t)lane * 16;
    v8h lo = *(const v8h*)(p);
    v8h hi = *(const v8h*)(p + 8);
    v16h b;
#pragma unroll
    for (int i = 0; i < 8; ++i) { b[i] = lo[i]; b[8 + i] = hi[i]; }
    return b;
}

// Fallback: B fragment gathered straight from row-major f32 global KxN weights.
__device__ __forceinline__ v16h load_b_global(const float* __restrict__ W, int ldn,
                                              int k0, int n0, int lane) {
    const int half = lane >> 4;
    const int c = lane & 15;
    const float* p = W + (size_t)(k0 + (half ? 8 : 0)) * ldn + (n0 + c);
    v16h b;
#pragma unroll
    for (int i = 0; i < 8; ++i) {
        b[i]     = (_Float16)p[(size_t)i * ldn];
        b[8 + i] = (_Float16)p[(size_t)(16 + i) * ldn];
    }
    return b;
}

// tanh + f32->f16 store of a D tile into a row-major f16 LDS buffer.
__device__ __forceinline__ void store_tanh_lds(_Float16* base, int ldn, int row0,
                                               int n0, int lane, v8f d) {
    const int half = lane >> 4;
    const int c = lane & 15;
#pragma unroll
    for (int i = 0; i < 8; ++i) {
        base[(size_t)(row0 + half * 8 + i) * ldn + n0 + c] = (_Float16)tanhf(d[i]);
    }
}

// ---------------------------------------------------------------------------
// Pass 1: convert all weights f32 -> f16 and swizzle into WMMA B-fragment
// layout. One thread packs one lane of one fragment (16 gathered f32 reads,
// one contiguous 32B write). ~5.6K fragments; trivial vs main pass.
// ---------------------------------------------------------------------------
__global__ __launch_bounds__(256) void pack_weights(
    const float* __restrict__ W1, const float* __restrict__ W2,
    const float* __restrict__ Wf, const float* __restrict__ Wg,
    _Float16* __restrict__ ws)
{
    const int gid  = blockIdx.x * blockDim.x + threadIdx.x;
    const int frag = gid >> 5;
    const int lane = gid & 31;

    const float* src; int ldn, ntN, ksN; size_t base; int local;
    if (frag < (int)F1) {
        local = frag;                       src = W1; ldn = kH; ntN = kH / 16; ksN = kS / 32; base = OFF_W1;
    } else if (frag < (int)(F1 + F2)) {
        local = frag - (int)F1;             src = W2; ldn = kH; ntN = kH / 16; ksN = kH / 32; base = OFF_W2;
    } else if (frag < (int)(F1 + F2 + Ff)) {
        local = frag - (int)(F1 + F2);      src = Wf; ldn = kS; ntN = kS / 16; ksN = kH / 32; base = OFF_Wf;
    } else if (frag < (int)(F1 + F2 + Ff + Fg)) {
        local = frag - (int)(F1 + F2 + Ff); src = Wg; ldn = kG; ntN = kG / 16; ksN = kH / 32; base = OFF_Wg;
    } else {
        return;
    }

    const int e  = local / (ntN * ksN);
    const int r  = local % (ntN * ksN);
    const int nt = r / ksN;
    const int ks = r % ksN;

    const float* We = src + (size_t)e * (size_t)(ksN * 32) * ldn;
    const int half = lane >> 4;
    const int c = lane & 15;
    const float* p = We + (size_t)(ks * 32 + (half ? 8 : 0)) * ldn + nt * 16 + c;

    v8h lo, hi;
#pragma unroll
    for (int i = 0; i < 8; ++i) {
        lo[i] = (_Float16)p[(size_t)i * ldn];
        hi[i] = (_Float16)p[(size_t)(16 + i) * ldn];
    }
    _Float16* dst = ws + base + (size_t)local * FRAG_H + (size_t)lane * 16;
    *(v8h*)(dst)     = lo;
    *(v8h*)(dst + 8) = hi;
}

// ---------------------------------------------------------------------------
// Pass 2: fused ensemble MLP. Each wave owns one N-column per layer and issues
// TWO WMMAs (M rows 0-15 and 16-31) per B fragment. Hot loops are explicitly
// double-buffered with sched_barrier fences so B-fragment loads stay a full
// iteration ahead of their consuming WMMAs. PACKED=false: small-ws fallback.
// ---------------------------------------------------------------------------
template <bool PACKED>
__global__ __launch_bounds__(THREADS, 1) void dyn_ensemble_fused(
    const float* __restrict__ state,
    const float* __restrict__ W1, const float* __restrict__ b1,
    const float* __restrict__ W2, const float* __restrict__ b2,
    const float* __restrict__ Wf, const float* __restrict__ bf,
    const float* __restrict__ Wg, const float* __restrict__ bg,
    const _Float16* __restrict__ wp,
    float* __restrict__ outF, float* __restrict__ outG)
{
    __shared__ _Float16 sX [BT * kS];   //  4 KB: state tile (f16)
    __shared__ _Float16 sH1[BT * kH];   // 16 KB: layer-1 activations
    __shared__ _Float16 sH2[BT * kH];   // 16 KB: layer-2 activations

    const int tid  = threadIdx.x;
    const int wave = tid >> 5;
    const int lane = tid & 31;
    const int col  = lane & 15;
    const int half = lane >> 4;
    const int rowBase = blockIdx.x * BT;

    // Stage state tile f32 -> f16 (one b128 load + one b64 LDS store per thread).
    static_assert(BT * kS / 4 == THREADS, "staging assumes one v4 per thread");
    {
        const v4f v = ((const v4f*)(state + (size_t)rowBase * kS))[tid];
        v4h h;
#pragma unroll
        for (int i = 0; i < 4; ++i) h[i] = (_Float16)v[i];
        ((v4h*)sX)[tid] = h;
    }
    __syncthreads();

    const int nW = wave;     // this wave's N-tile for the 256-wide layers

    const _Float16* w1p = PACKED ? wp + OFF_W1 : nullptr;
    const _Float16* w2p = PACKED ? wp + OFF_W2 : nullptr;
    const _Float16* wfp = PACKED ? wp + OFF_Wf : nullptr;
    const _Float16* wgp = PACKED ? wp + OFF_Wg : nullptr;

    v8f accF[2];             // f head accumulators (waves 0..3), M-tiles 0/1
    v8f accG[4][2];          // g head: 4 N-tiles x 2 M-tiles per wave
#pragma unroll
    for (int m = 0; m < 2; ++m)
#pragma unroll
        for (int i = 0; i < 8; ++i) accF[m][i] = 0.0f;
#pragma unroll
    for (int t = 0; t < 4; ++t)
#pragma unroll
        for (int m = 0; m < 2; ++m)
#pragma unroll
            for (int i = 0; i < 8; ++i) accG[t][m][i] = 0.0f;

    for (int e = 0; e < kE; ++e) {
        // ---- Layer 1: h1 = tanh(x @ W1e + b1e). K = 64; one N-tile, both M.
        {
            const int n0 = nW * 16;
            const float bias = b1[(size_t)e * kH + n0 + col];
            v8f c0, c1;
#pragma unroll
            for (int i = 0; i < 8; ++i) { c0[i] = bias; c1[i] = bias; }

            v16h b0, b1f;
            if constexpr (PACKED) {
                b0  = load_b_packed(w1p, (size_t)(e * 16 + nW) * 2 + 0, lane);
                b1f = load_b_packed(w1p, (size_t)(e * 16 + nW) * 2 + 1, lane);
            } else {
                b0  = load_b_global(W1 + (size_t)e * kS * kH, kH, 0,  n0, lane);
                b1f = load_b_global(W1 + (size_t)e * kS * kH, kH, 32, n0, lane);
            }
            v16h a00 = load_a_lds(sX, kS, 0,  0,  lane);
            v16h a10 = load_a_lds(sX, kS, 16, 0,  lane);
            v16h a01 = load_a_lds(sX, kS, 0,  32, lane);
            v16h a11 = load_a_lds(sX, kS, 16, 32, lane);
            SCHED_FENCE();
            c0 = wmma_f16(a00, b0,  c0);
            c1 = wmma_f16(a10, b0,  c1);
            c0 = wmma_f16(a01, b1f, c0);
            c1 = wmma_f16(a11, b1f, c1);
            store_tanh_lds(sH1, kH, 0,  n0, lane, c0);
            store_tanh_lds(sH1, kH, 16, n0, lane, c1);
        }
        __syncthreads();  // h1 ready; fences prior-member head reads of sH2

        // ---- Layer 2: h2 = tanh(h1 @ W2e + b2e). K = 256; distance-2 pipeline.
        {
            const int n0 = nW * 16;
            const float bias = b2[(size_t)e * kH + n0 + col];
            v8f c0, c1;
#pragma unroll
            for (int i = 0; i < 8; ++i) { c0[i] = bias; c1[i] = bias; }

            if constexpr (PACKED) {
                const size_t base2 = (size_t)(e * 16 + nW) * 8;
                v16h buf[2];
                buf[0] = load_b_packed(w2p, base2 + 0, lane);
                buf[1] = load_b_packed(w2p, base2 + 1, lane);
#pragma unroll
                for (int ks = 0; ks < 8; ++ks) {
                    v16h bc = buf[ks & 1];
                    if (ks + 2 < 8)
                        buf[ks & 1] = load_b_packed(w2p, base2 + ks + 2, lane);
                    v16h a0 = load_a_lds(sH1, kH, 0,  ks * 32, lane);
                    v16h a1 = load_a_lds(sH1, kH, 16, ks * 32, lane);
                    SCHED_FENCE();   // keep prefetch ahead of this ks's WMMAs
                    c0 = wmma_f16(a0, bc, c0);
                    c1 = wmma_f16(a1, bc, c1);
                }
            } else {
                const float* W2e = W2 + (size_t)e * kH * kH;
#pragma unroll
                for (int ks = 0; ks < 8; ++ks) {
                    v16h b = load_b_global(W2e, kH, ks * 32, n0, lane);
                    v16h a0 = load_a_lds(sH1, kH, 0,  ks * 32, lane);
                    v16h a1 = load_a_lds(sH1, kH, 16, ks * 32, lane);
                    c0 = wmma_f16(a0, b, c0);
                    c1 = wmma_f16(a1, b, c1);
                }
            }
            store_tanh_lds(sH2, kH, 0,  n0, lane, c0);
            store_tanh_lds(sH2, kH, 16, n0, lane, c1);
        }
        __syncthreads();  // h2 ready before heads read it

        // ---- f head: 4 N-tiles on waves 0..3, both M per wave.
        if (wave < 4) {
            if constexpr (PACKED) {
                const size_t basef = (size_t)(e * 4 + nW) * 8;
                v16h buf[2];
                buf[0] = load_b_packed(wfp, basef + 0, lane);
                buf[1] = load_b_packed(wfp, basef + 1, lane);
#pragma unroll
                for (int ks = 0; ks < 8; ++ks) {
                    v16h bc = buf[ks & 1];
                    if (ks + 2 < 8)
                        buf[ks & 1] = load_b_packed(wfp, basef + ks + 2, lane);
                    v16h a0 = load_a_lds(sH2, kH, 0,  ks * 32, lane);
                    v16h a1 = load_a_lds(sH2, kH, 16, ks * 32, lane);
                    SCHED_FENCE();
                    accF[0] = wmma_f16(a0, bc, accF[0]);
                    accF[1] = wmma_f16(a1, bc, accF[1]);
                }
            } else {
                const float* Wfe = Wf + (size_t)e * kH * kS;
#pragma unroll
                for (int ks = 0; ks < 8; ++ks) {
                    v16h b = load_b_global(Wfe, kS, ks * 32, nW * 16, lane);
                    v16h a0 = load_a_lds(sH2, kH, 0,  ks * 32, lane);
                    v16h a1 = load_a_lds(sH2, kH, 16, ks * 32, lane);
                    accF[0] = wmma_f16(a0, b, accF[0]);
                    accF[1] = wmma_f16(a1, b, accF[1]);
                }
            }
        }

        if (e + 1 < kE) {  // warm caches for the next member's big weights
            if constexpr (PACKED) {
                __builtin_prefetch(w2p + (size_t)(e + 1) * 16 * 8 * FRAG_H, 0, 0);
                __builtin_prefetch(wgp + (size_t)(e + 1) * 64 * 8 * FRAG_H, 0, 0);
            } else {
                __builtin_prefetch(W2 + (size_t)(e + 1) * kH * kH, 0, 0);
                __builtin_prefetch(Wg + (size_t)(e + 1) * kH * kG, 0, 0);
            }
        }

        // ---- g head: wave owns N-tiles nW+16t (t=0..3), both M per fragment.
        // frag(t,ks) = (e*64 + nW + 16t)*8 + ks = baseg + t*128 + ks.
        // ks-level double buffer: while WMMAing group ks, the 4 fragments of
        // ks+1 (8 b128 loads) are in flight; sched fence stops load sinking.
        if constexpr (PACKED) {
            const size_t baseg = (size_t)(e * 64 + nW) * 8;
            v16h buf[8];                       // [parity*4 + t]
#pragma unroll
            for (int t = 0; t < 4; ++t)
                buf[t] = load_b_packed(wgp, baseg + (size_t)t * 128, lane);
#pragma unroll
            for (int ks = 0; ks < 8; ++ks) {
                const int cb = (ks & 1) * 4;   // current group
                const int nb = 4 - cb;         // next group
                if (ks < 7) {
#pragma unroll
                    for (int t = 0; t < 4; ++t)
                        buf[nb + t] = load_b_packed(
                            wgp, baseg + (size_t)t * 128 + ks + 1, lane);
                }
                v16h a0 = load_a_lds(sH2, kH, 0,  ks * 32, lane);
                v16h a1 = load_a_lds(sH2, kH, 16, ks * 32, lane);
                SCHED_FENCE();                 // all loads above, WMMAs below
#pragma unroll
                for (int t = 0; t < 4; ++t) {
                    accG[t][0] = wmma_f16(a0, buf[cb + t], accG[t][0]);
                    accG[t][1] = wmma_f16(a1, buf[cb + t], accG[t][1]);
                }
            }
        } else {
            const float* Wge = Wg + (size_t)e * kH * kG;
#pragma unroll
            for (int ks = 0; ks < 8; ++ks) {
                v16h a0 = load_a_lds(sH2, kH, 0,  ks * 32, lane);
                v16h a1 = load_a_lds(sH2, kH, 16, ks * 32, lane);
#pragma unroll
                for (int t = 0; t < 4; ++t) {
                    v16h b = load_b_global(Wge, kG, ks * 32, (nW + 16 * t) * 16, lane);
                    accG[t][0] = wmma_f16(a0, b, accG[t][0]);
                    accG[t][1] = wmma_f16(a1, b, accG[t][1]);
                }
            }
        }
    }

    // ---- Epilogue: ensemble mean (sum + summed bias, times 1/8), f32 out.
    if (wave < 4) {
        const int n0 = nW * 16;
        float bsum = 0.0f;
#pragma unroll
        for (int e = 0; e < kE; ++e) bsum += bf[(size_t)e * kS + n0 + col];
#pragma unroll
        for (int m = 0; m < 2; ++m)
#pragma unroll
            for (int i = 0; i < 8; ++i) {
                const int row = rowBase + m * 16 + half * 8 + i;
                outF[(size_t)row * kS + n0 + col] = (accF[m][i] + bsum) * 0.125f;
            }
    }
#pragma unroll
    for (int t = 0; t < 4; ++t) {
        const int n0 = (nW + 16 * t) * 16;
        float bsum = 0.0f;
#pragma unroll
        for (int e = 0; e < kE; ++e) bsum += bg[(size_t)e * kG + n0 + col];
#pragma unroll
        for (int m = 0; m < 2; ++m)
#pragma unroll
            for (int i = 0; i < 8; ++i) {
                const int row = rowBase + m * 16 + half * 8 + i;
                outG[(size_t)row * kG + n0 + col] = (accG[t][m][i] + bsum) * 0.125f;
            }
    }
}

extern "C" void kernel_launch(void* const* d_in, const int* in_sizes, int n_in,
                              void* d_out, int out_size, void* d_ws, size_t ws_size,
                              hipStream_t stream) {
    const float* state = (const float*)d_in[0];
    const float* W1 = (const float*)d_in[1];
    const float* b1 = (const float*)d_in[2];
    const float* W2 = (const float*)d_in[3];
    const float* b2 = (const float*)d_in[4];
    const float* Wf = (const float*)d_in[5];
    const float* bf = (const float*)d_in[6];
    const float* Wg = (const float*)d_in[7];
    const float* bg = (const float*)d_in[8];

    const int nBatch = in_sizes[0] / kS;        // 16384
    float* outF = (float*)d_out;                // [B, S]
    float* outG = outF + (size_t)nBatch * kS;   // [B, S*A] (flat == [B,S,A])

    dim3 grid(nBatch / BT), block(THREADS);

    if (ws_size >= NEED_WS_BYTES) {
        _Float16* wp = (_Float16*)d_ws;
        const int packWork = (int)((F1 + F2 + Ff + Fg) * 32);
        hipLaunchKernelGGL(pack_weights, dim3((packWork + 255) / 256), dim3(256), 0,
                           stream, W1, W2, Wf, Wg, wp);
        hipLaunchKernelGGL((dyn_ensemble_fused<true>), grid, block, 0, stream,
                           state, W1, b1, W2, b2, Wf, bf, Wg, bg, wp, outF, outG);
    } else {
        hipLaunchKernelGGL((dyn_ensemble_fused<false>), grid, block, 0, stream,
                           state, W1, b1, W2, b2, Wf, bf, Wg, bg,
                           (const _Float16*)nullptr, outF, outG);
    }
}
